// SELFContrastiveLoss_43800076485241
// MI455X (gfx1250) — compile-verified, matrix operand order checked by reading, not run on hardware
//
#include <hip/hip_runtime.h>
#include <hip/hip_bf16.h>

// ---- problem constants (match reference) ----
#define BATCH   8192
#define DIM     512
#define INV_T   2.0f       // 1/TEMP, TEMP=0.5
#define EPS     1e-5f
#define NORMEPS 1e-12f

typedef __attribute__((ext_vector_type(16))) _Float16 v16h;
typedef __attribute__((ext_vector_type(8)))  _Float16 v8h;
typedef __attribute__((ext_vector_type(4)))  _Float16 v4h;
typedef __attribute__((ext_vector_type(8)))  float    v8f;
typedef __attribute__((ext_vector_type(4)))  float    v4f;

// ---------------------------------------------------------------------------
// Kernel 1: row L2-normalize q and k (fp32) -> f16 tiles in workspace.
// One wave per row; 2*BATCH rows, 8 waves per block.
// ---------------------------------------------------------------------------
__global__ __launch_bounds__(256) void norm_f16_kernel(const float* __restrict__ q,
                                                       const float* __restrict__ k,
                                                       _Float16* __restrict__ qh,
                                                       _Float16* __restrict__ kh) {
    const int lane = threadIdx.x & 31;
    const int row  = blockIdx.x * 8 + (threadIdx.x >> 5);
    const float* src;
    _Float16*    dst;
    if (row < BATCH) { src = q + (size_t)row * DIM;           dst = qh + (size_t)row * DIM; }
    else             { src = k + (size_t)(row - BATCH) * DIM; dst = kh + (size_t)(row - BATCH) * DIM; }

    v4f x[4];
    float ss = 0.0f;
#pragma unroll
    for (int i = 0; i < 4; ++i) {
        x[i] = *(const v4f*)(src + lane * 4 + i * 128);
#pragma unroll
        for (int j = 0; j < 4; ++j) ss += x[i][j] * x[i][j];
    }
#pragma unroll
    for (int m = 1; m < 32; m <<= 1) ss += __shfl_xor(ss, m, 32);

    const float inv = 1.0f / fmaxf(sqrtf(ss), NORMEPS);
#pragma unroll
    for (int i = 0; i < 4; ++i) {
        v4h h;
#pragma unroll
        for (int j = 0; j < 4; ++j) h[j] = (_Float16)(x[i][j] * inv);
        *(v4h*)(dst + lane * 4 + i * 128) = h;
    }
}

// ---------------------------------------------------------------------------
// Kernel 2: zero the rowsum/colsum accumulators (contiguous 2*BATCH floats).
// ---------------------------------------------------------------------------
__global__ void zero_kernel(float* __restrict__ p, int n) {
    int i = blockIdx.x * 256 + threadIdx.x;
    if (i < n) p[i] = 0.0f;
}

// ---------------------------------------------------------------------------
// Kernel 3: positives d[i] = qn[i] . kn[i] over the SAME f16 data the GEMM
// consumes (rounding-consistent with the row/col sums). One wave per row.
// ---------------------------------------------------------------------------
__global__ __launch_bounds__(256) void epos_kernel(const _Float16* __restrict__ qh,
                                                   const _Float16* __restrict__ kh,
                                                   float* __restrict__ diag) {
    const int lane = threadIdx.x & 31;
    const int row  = blockIdx.x * 8 + (threadIdx.x >> 5);
    const _Float16* a = qh + (size_t)row * DIM + lane * 16;
    const _Float16* b = kh + (size_t)row * DIM + lane * 16;
    const v8h a0 = *(const v8h*)a, a1 = *(const v8h*)(a + 8);
    const v8h b0 = *(const v8h*)b, b1 = *(const v8h*)(b + 8);
    float s = 0.0f;
#pragma unroll
    for (int j = 0; j < 8; ++j)
        s += (float)a0[j] * (float)b0[j] + (float)a1[j] * (float)b1[j];
#pragma unroll
    for (int m = 1; m < 32; m <<= 1) s += __shfl_xor(s, m, 32);
    if (lane == 0) diag[row] = s;
}

// ---------------------------------------------------------------------------
// Kernel 4: fused  S = qn @ kn^T  ->  E = exp(2*S), full row/col sums.
// No diagonal masking needed: denom_qk = e_pos + offdiag-rowsum == FULL row
// sum of E (E[i,i] == e_pos), so the epilogue is branch-free.
// Block = 256 threads = 8 waves; block tile = 256x128; wave tile = 64x64
// (4x4 fragments -> 16 b128 loads per 16 WMMAs, 2x the arithmetic intensity
// of a 64x32 tile, to keep the XDL fed from the CU load path).
// ---------------------------------------------------------------------------
__global__ __launch_bounds__(256) void gemm_softmax_kernel(const _Float16* __restrict__ qh,
                                                           const _Float16* __restrict__ kh,
                                                           float* __restrict__ rowsum,
                                                           float* __restrict__ colsum) {
    const int lane   = threadIdx.x & 31;
    const int wid    = threadIdx.x >> 5;
    const int waveM  = wid >> 1;          // 0..3  (64-row band of 256)
    const int waveN  = wid & 1;           // 0..1  (64-col band of 128)
    const int lane15 = lane & 15;
    const int khalf  = lane >> 4;         // ISA 16-bit A/B layout: K-octet select
    const int tM = blockIdx.x, tN = blockIdx.y;
    const int rowBase = tM * 256 + waveM * 64;
    const int colBase = tN * 128 + waveN * 64;

    __shared__ float ldsRow[256];
    __shared__ float ldsCol[128];
    ldsRow[threadIdx.x] = 0.0f;
    if (threadIdx.x < 128) ldsCol[threadIdx.x] = 0.0f;
    __syncthreads();

    // Per-lane base pointers matching the 16-bit A/B VGPR layout:
    // lane%16 = matrix row, lane/16 selects the K-octet within each 16-group.
    const _Float16* aBase[4];
    const _Float16* bBase[4];
#pragma unroll
    for (int mi = 0; mi < 4; ++mi)
        aBase[mi] = qh + (size_t)(rowBase + mi * 16 + lane15) * DIM + khalf * 8;
#pragma unroll
    for (int ni = 0; ni < 4; ++ni)
        bBase[ni] = kh + (size_t)(colBase + ni * 16 + lane15) * DIM + khalf * 8;

    v8f acc[4][4];
    const v8f zf = {0.f, 0.f, 0.f, 0.f, 0.f, 0.f, 0.f, 0.f};
#pragma unroll
    for (int mi = 0; mi < 4; ++mi)
#pragma unroll
        for (int ni = 0; ni < 4; ++ni) acc[mi][ni] = zf;

    for (int kk = 0; kk < DIM; kk += 32) {
        // prefetch next K chunk (speculative; dropped past end)
        __builtin_prefetch(aBase[0] + kk + 32, 0, 3);
        __builtin_prefetch(bBase[0] + kk + 32, 0, 3);

        v16h a[4], b[4];
#pragma unroll
        for (int mi = 0; mi < 4; ++mi) {
            union { v16h v; v8h h[2]; } u;
            u.h[0] = *(const v8h*)(aBase[mi] + kk);        // K = khalf*8 .. +7
            u.h[1] = *(const v8h*)(aBase[mi] + kk + 16);   // K = 16+khalf*8 .. +7
            a[mi] = u.v;
        }
#pragma unroll
        for (int ni = 0; ni < 4; ++ni) {
            union { v16h v; v8h h[2]; } u;
            u.h[0] = *(const v8h*)(bBase[ni] + kk);
            u.h[1] = *(const v8h*)(bBase[ni] + kk + 16);
            b[ni] = u.v;
        }
#pragma unroll
        for (int mi = 0; mi < 4; ++mi)
#pragma unroll
            for (int ni = 0; ni < 4; ++ni)
                acc[mi][ni] = __builtin_amdgcn_wmma_f32_16x16x32_f16(
                    false, a[mi], false, b[ni], (short)0, acc[mi][ni], false, false);
    }

    // Branch-free epilogue: E = exp(2*S), accumulate row/col partials.
    // C/D layout: lane%16 = N, VGPR r holds M = r + 8*(lane/16).
    float rsum[4][8];
    float csum[4] = {0.0f, 0.0f, 0.0f, 0.0f};
#pragma unroll
    for (int mi = 0; mi < 4; ++mi)
#pragma unroll
        for (int r = 0; r < 8; ++r) rsum[mi][r] = 0.0f;

#pragma unroll
    for (int mi = 0; mi < 4; ++mi)
#pragma unroll
        for (int ni = 0; ni < 4; ++ni)
#pragma unroll
            for (int r = 0; r < 8; ++r) {
                const float e = __expf(acc[mi][ni][r] * INV_T);
                rsum[mi][r] += e;
                csum[ni]    += e;
            }

    // column partials: lanes L and L+16 cover complementary rows, same column
#pragma unroll
    for (int ni = 0; ni < 4; ++ni)
        atomicAdd(&ldsCol[waveN * 64 + ni * 16 + lane15], csum[ni]);

    // row partials: butterfly over the 16-lane half group (masks < 16 keep the
    // lane/16 half invariant), then lanes 0 and 16 flush their 8 rows to LDS
#pragma unroll
    for (int mi = 0; mi < 4; ++mi) {
#pragma unroll
        for (int r = 0; r < 8; ++r) {
            float v = rsum[mi][r];
            v += __shfl_xor(v, 1, 32);
            v += __shfl_xor(v, 2, 32);
            v += __shfl_xor(v, 4, 32);
            v += __shfl_xor(v, 8, 32);
            rsum[mi][r] = v;
        }
        if (lane15 == 0) {
#pragma unroll
            for (int r = 0; r < 8; ++r)
                atomicAdd(&ldsRow[waveM * 64 + mi * 16 + (khalf << 3) + r], rsum[mi][r]);
        }
    }
    __syncthreads();

    atomicAdd(&rowsum[tM * 256 + threadIdx.x], ldsRow[threadIdx.x]);
    if (threadIdx.x < 128)
        atomicAdd(&colsum[tN * 128 + threadIdx.x], ldsCol[threadIdx.x]);
}

// ---------------------------------------------------------------------------
// Kernel 5: final scalar loss. Denominators are the FULL row/col sums.
// ---------------------------------------------------------------------------
__global__ __launch_bounds__(256) void loss_kernel(const float* __restrict__ diag,
                                                   const float* __restrict__ rowsum,
                                                   const float* __restrict__ colsum,
                                                   float* __restrict__ out) {
    __shared__ float red[256];
    float acc = 0.0f;
    for (int i = threadIdx.x; i < BATCH; i += 256) {
        const float ep = __expf(diag[i] * INV_T);
        acc += -__logf(ep / rowsum[i] + EPS)
             -  __logf(ep / colsum[i] + EPS);
    }
    red[threadIdx.x] = acc;
    __syncthreads();
    for (int s = 128; s > 0; s >>= 1) {
        if (threadIdx.x < s) red[threadIdx.x] += red[threadIdx.x + s];
        __syncthreads();
    }
    if (threadIdx.x == 0) out[0] = red[0] / (float)BATCH;
}

// ---------------------------------------------------------------------------
extern "C" void kernel_launch(void* const* d_in, const int* in_sizes, int n_in,
                              void* d_out, int out_size, void* d_ws, size_t ws_size,
                              hipStream_t stream) {
    const float* q = (const float*)d_in[0];
    const float* k = (const float*)d_in[1];
    float* out = (float*)d_out;

    const size_t HALF_BYTES = (size_t)BATCH * DIM * sizeof(_Float16);   // 8 MB each
    char* ws = (char*)d_ws;
    _Float16* qh     = (_Float16*)ws;
    _Float16* kh     = (_Float16*)(ws + HALF_BYTES);
    float*    rowsum = (float*)(ws + 2 * HALF_BYTES);
    float*    colsum = rowsum + BATCH;
    float*    diag   = colsum + BATCH;

    // 1) normalize + convert: 2*BATCH rows, 8 rows/block
    norm_f16_kernel<<<(2 * BATCH) / 8, 256, 0, stream>>>(q, k, qh, kh);
    // 2) zero accumulators (rowsum|colsum contiguous)
    zero_kernel<<<(2 * BATCH + 255) / 256, 256, 0, stream>>>(rowsum, 2 * BATCH);
    // 3) positives (diag of S) from the same f16 data
    epos_kernel<<<BATCH / 8, 256, 0, stream>>>(qh, kh, diag);
    // 4) fused WMMA GEMM + exp + full row/col reductions (branch-free)
    dim3 grid(BATCH / 256, BATCH / 128);
    gemm_softmax_kernel<<<grid, 256, 0, stream>>>(qh, kh, rowsum, colsum);
    // 5) scalar loss
    loss_kernel<<<1, 256, 0, stream>>>(diag, rowsum, colsum, out);
}
